// LSTM_37099927503217
// MI455X (gfx1250) — matrix operand order repaired and checked by reading, not compiled, not run on hardware
//
#include <hip/hip_runtime.h>

#define B_ 64
#define T_ 1024
#define H_ 512
#define G_ 2048   // 4*H
#define K_ 512
#define M_ (B_*T_) // 65536
#define WSTRIDE 520  // Wh LDS column stride (bf16): 260 dwords == 4 mod 64
                     // -> 16 lanes x 4-dword ds_load_b128 tile all 64 banks

typedef __attribute__((ext_vector_type(16))) __bf16 v16bf;
typedef __attribute__((ext_vector_type(8)))  __bf16 v8bf;
typedef __attribute__((ext_vector_type(8)))  float  v8f;

__device__ __forceinline__ __bf16 f2bf(float f) {
  unsigned u = __builtin_bit_cast(unsigned, f);
  u += 0x7FFFu + ((u >> 16) & 1u);               // round-to-nearest-even
  unsigned short h = (unsigned short)(u >> 16);
  return __builtin_bit_cast(__bf16, h);
}

__device__ __forceinline__ float sigm(float x) { return 1.0f / (1.0f + __expf(-x)); }

// ---------------------------------------------------------------------------
// Convert fp32 W[K_, G_] (row-major, N-contiguous) -> bf16 WT[G_, K_]
// (K-contiguous) so each WMMA B-fragment lane reads two contiguous 16B runs.
// ---------------------------------------------------------------------------
__global__ void k_wt_bf16(const float* __restrict__ W, __bf16* __restrict__ WT) {
  int idx = blockIdx.x * blockDim.x + threadIdx.x; // K_*G_ = 2^20 total
  int n = idx & (G_ - 1);
  int k = idx >> 11;                               // /G_ (2048)
  WT[(size_t)n * K_ + k] = f2bf(W[idx]);
}

// ---------------------------------------------------------------------------
// XZ[M, G_] = A[M, K_] * W + bias.  A is fp32 (layer 0 input) or bf16 (y0).
// Block = 128 threads (4 waves). Wave computes a 16(M) x 64(N) strip as
// 4 WMMA tiles; block covers 64x64. bf16 WMMA, fp32 accumulate.
//
// Fragment layouts per CDNA5 ISA 7.12.2 (wave32):
//  A 16x32 bf16: lanes 0-15 row M=lane, K = 0..7 & 16..23; lanes 16-31 same
//                rows, K = 8..15 & 24..31.
//  B 32x16 bf16: lane n = column, lanes 0-15 K=0..7 & 16..23, lanes 16-31 K=8..15 & 24..31.
//  C/D 16x16 f32: VGPR r -> row r (lanes 0-15) / row r+8 (lanes 16-31),
//                col = lane&15.
// ---------------------------------------------------------------------------
template<bool A_FP32>
__global__ void k_gemm_xz(const void* __restrict__ Aptr,
                          const __bf16* __restrict__ WT,
                          const float* __restrict__ bias,
                          float* __restrict__ XZ) {
  const int lane    = threadIdx.x & 31;
  const int wave    = threadIdx.x >> 5;
  const int m0      = blockIdx.x * 64 + wave * 16;
  const int n0      = blockIdx.y * 64;
  const int halfsel = (lane >> 4) & 1;
  const int rowA    = m0 + (lane & 15);
  const int cA      = halfsel ? 8 : 0;
  const int kkB     = halfsel ? 8 : 0;

  v8f acc[4];
#pragma unroll
  for (int i = 0; i < 4; ++i)
#pragma unroll
    for (int j = 0; j < 8; ++j) acc[i][j] = 0.0f;

  for (int k0 = 0; k0 < K_; k0 += 32) {
    v16bf afrag;
    if (A_FP32) {
      const float* ap = (const float*)Aptr + (size_t)rowA * K_ + k0 + cA;
      float4 x0 = *(const float4*)(ap + 0);
      float4 x1 = *(const float4*)(ap + 4);
      float4 x2 = *(const float4*)(ap + 16);
      float4 x3 = *(const float4*)(ap + 20);
      afrag[0]  = f2bf(x0.x); afrag[1]  = f2bf(x0.y); afrag[2]  = f2bf(x0.z); afrag[3]  = f2bf(x0.w);
      afrag[4]  = f2bf(x1.x); afrag[5]  = f2bf(x1.y); afrag[6]  = f2bf(x1.z); afrag[7]  = f2bf(x1.w);
      afrag[8]  = f2bf(x2.x); afrag[9]  = f2bf(x2.y); afrag[10] = f2bf(x2.z); afrag[11] = f2bf(x2.w);
      afrag[12] = f2bf(x3.x); afrag[13] = f2bf(x3.y); afrag[14] = f2bf(x3.z); afrag[15] = f2bf(x3.w);
    } else {
      const __bf16* ap = (const __bf16*)Aptr + (size_t)rowA * K_ + k0 + cA;
      v8bf lo = *(const v8bf*)ap;
      v8bf hi = *(const v8bf*)(ap + 16);
#pragma unroll
      for (int j = 0; j < 8; ++j) { afrag[j] = lo[j]; afrag[j + 8] = hi[j]; }
    }
#pragma unroll
    for (int nt = 0; nt < 4; ++nt) {
      const __bf16* bp = WT + (size_t)(n0 + nt * 16 + (lane & 15)) * K_ + k0 + kkB;
      __builtin_prefetch(bp + 32, 0, 3);   // global_prefetch_b8: next K tile
      v8bf blo = *(const v8bf*)bp;
      v8bf bhi = *(const v8bf*)(bp + 16);
      v16bf bfrag;
#pragma unroll
      for (int j = 0; j < 8; ++j) { bfrag[j] = blo[j]; bfrag[j + 8] = bhi[j]; }
      acc[nt] = __builtin_amdgcn_wmma_f32_16x16x32_bf16(
          false, afrag, false, bfrag, (short)0, acc[nt], false, false);
    }
  }

  const int crow = m0 + (halfsel ? 8 : 0);
  const int ccol = n0 + (lane & 15);
#pragma unroll
  for (int nt = 0; nt < 4; ++nt) {
    int col = ccol + nt * 16;
    float bb = bias[col];
#pragma unroll
    for (int r = 0; r < 8; ++r)
      XZ[(size_t)(crow + r) * G_ + col] = acc[nt][r] + bb;
  }
}

// ---------------------------------------------------------------------------
// Persistent recurrence kernel. 32 blocks x 128 threads. Block owns hidden
// slice j0 = blockIdx.x*16. Wave w owns batch tile mt=w (16 unique rows) and
// computes all 4 gate columns for it, so h A-fragments are loaded exactly
// once per wave. The block's time-invariant Wh^T slice (64 cols x 512 K bf16)
// is staged into LDS ONCE via global_load_async_to_lds_b128 (ASYNCcnt) with
// a padded column stride (WSTRIDE=520) so ds_load_b128 B-fragment reads are
// bank-conflict-free, and is served from LDS for all 1024 steps. Gate
// exchange via LDS, bf16 h double-buffered in global, device-wide barrier
// per step, and xz pre-activations for step t+1 prefetched during step t.
// Dynamic LDS: 64*520*2 + 16 KB ~= 81 KB (WGP has 320 KB).
// ---------------------------------------------------------------------------
template<bool LAST>
__global__ void k_lstm_recur(const float* __restrict__ xz,
                             const __bf16* __restrict__ WhT,
                             __bf16* __restrict__ hbuf,   // [2][B_][H_] bf16
                             float* __restrict__ cst,     // [B_][H_] fp32
                             __bf16* __restrict__ ybf,    // layer-0 output
                             float*  __restrict__ yf32,   // layer-1 output
                             float* __restrict__ hfin,
                             float* __restrict__ cfin,
                             unsigned* __restrict__ cnt) {
  extern __shared__ __align__(16) char smem_dyn[];
  __bf16* whlds = (__bf16*)smem_dyn;                       // [64][WSTRIDE] bf16
  typedef float zsh_row[64][16];
  zsh_row* zsh = (zsh_row*)(smem_dyn + (size_t)64 * WSTRIDE * 2); // [4][64][16]

  const int lane    = threadIdx.x & 31;
  const int wave    = threadIdx.x >> 5;            // batch tile 0..3
  const int j0      = blockIdx.x * 16;
  const int halfsel = (lane >> 4) & 1;
  const int cA      = halfsel ? 8 : 0;
  const int kkB     = halfsel ? 8 : 0;
  const unsigned nb = gridDim.x;

  // ---- one-time async stage of Wh^T slice into LDS (ASYNCcnt tracked) ----
  {
#pragma unroll 1
    for (int it = 0; it < 32; ++it) {
      const int c   = it * 128 + (int)threadIdx.x; // 16B chunk id, 0..4095
      const int col = c >> 6;                      // LDS column 0..63
      const int cc  = c & 63;                      // 16B chunk within column
      const int gate = col >> 4, jj = col & 15;
      const __bf16* gsrc = WhT + (size_t)(gate * H_ + j0 + jj) * K_ + (size_t)cc * 8;
      unsigned lds_addr =
          (unsigned)(uintptr_t)(whlds + (size_t)col * WSTRIDE + (size_t)cc * 8);
      unsigned long long ga = (unsigned long long)(uintptr_t)gsrc;
      asm volatile("global_load_async_to_lds_b128 %0, %1, off"
                   :: "v"(lds_addr), "v"(ga) : "memory");
    }
    asm volatile("s_wait_asynccnt 0x0" ::: "memory");
    __syncthreads();
  }

#pragma unroll 1
  for (int t = 0; t < T_; ++t) {
    const __bf16* hread  = hbuf + (size_t)(t & 1) * (B_ * H_);
    __bf16*       hwrite = hbuf + (size_t)((t + 1) & 1) * (B_ * H_);

    v8f acc[4];
#pragma unroll
    for (int i = 0; i < 4; ++i)
#pragma unroll
      for (int j = 0; j < 8; ++j) acc[i][j] = 0.0f;

    const int rowA = wave * 16 + (lane & 15);      // unique batch row per wave
    for (int k0 = 0; k0 < K_; k0 += 32) {
      const __bf16* ap = hread + (size_t)rowA * H_ + k0 + cA;
      v8bf alo = *(const v8bf*)ap;
      v8bf ahi = *(const v8bf*)(ap + 16);
      v16bf afrag;
#pragma unroll
      for (int j = 0; j < 8; ++j) { afrag[j] = alo[j]; afrag[j + 8] = ahi[j]; }
#pragma unroll
      for (int nt = 0; nt < 4; ++nt) {             // nt = gate
        const __bf16* bp =
            whlds + (size_t)(nt * 16 + (lane & 15)) * WSTRIDE + k0 + kkB;
        v8bf blo = *(const v8bf*)bp;               // ds_load_b128, no conflicts
        v8bf bhi = *(const v8bf*)(bp + 16);
        v16bf bfrag;
#pragma unroll
        for (int j = 0; j < 8; ++j) { bfrag[j] = blo[j]; bfrag[j + 8] = bhi[j]; }
        acc[nt] = __builtin_amdgcn_wmma_f32_16x16x32_bf16(
            false, afrag, false, bfrag, (short)0, acc[nt], false, false);
      }
    }

    const int brow = wave * 16 + (halfsel ? 8 : 0);
#pragma unroll
    for (int nt = 0; nt < 4; ++nt) {
      const int colG = nt * H_ + j0 + (lane & 15);
#pragma unroll
      for (int r = 0; r < 8; ++r) {
        const size_t xidx = ((size_t)(brow + r) * T_ + t) * G_ + colG;
        float z = acc[nt][r] + xz[xidx];
        zsh[nt][brow + r][lane & 15] = z;
        __builtin_prefetch(&xz[xidx + G_], 0, 3); // pull step t+1 slice early
      }
    }
    __syncthreads();

    for (int idx = threadIdx.x; idx < B_ * 16; idx += 128) {
      const int b  = idx >> 4;
      const int jj = idx & 15;
      const float iv = sigm(zsh[0][b][jj]);
      const float fv = sigm(zsh[1][b][jj]);
      const float gv = tanhf(zsh[2][b][jj]);
      const float ov = sigm(zsh[3][b][jj]);
      const size_t sidx = (size_t)b * H_ + j0 + jj;
      const float cnew = fv * cst[sidx] + iv * gv;
      const float hnew = ov * tanhf(cnew);
      cst[sidx]    = cnew;
      hwrite[sidx] = f2bf(hnew);
      const size_t yidx = ((size_t)b * T_ + t) * H_ + j0 + jj;
      if (LAST) yf32[yidx] = hnew; else ybf[yidx] = f2bf(hnew);
      if (t == T_ - 1) { hfin[sidx] = hnew; cfin[sidx] = cnew; }
    }
    __syncthreads();
    __threadfence();                               // publish h slice, DEV scope
    if (threadIdx.x == 0) {
      __hip_atomic_fetch_add(cnt, 1u, __ATOMIC_RELEASE, __HIP_MEMORY_SCOPE_AGENT);
      const unsigned target = nb * (unsigned)(t + 1);
      while (__hip_atomic_load(cnt, __ATOMIC_ACQUIRE, __HIP_MEMORY_SCOPE_AGENT) < target)
        __builtin_amdgcn_s_sleep(1);
    }
    __syncthreads();
    __threadfence();                               // invalidate stale WGP$
  }
}

// ---------------------------------------------------------------------------
extern "C" void kernel_launch(void* const* d_in, const int* in_sizes, int n_in,
                              void* d_out, int out_size, void* d_ws, size_t ws_size,
                              hipStream_t stream) {
  (void)in_sizes; (void)n_in; (void)out_size; (void)ws_size;
  const float* x   = (const float*)d_in[0];
  const float* Wx0 = (const float*)d_in[1];
  const float* Wh0 = (const float*)d_in[2];
  const float* b0  = (const float*)d_in[3];
  const float* Wx1 = (const float*)d_in[4];
  const float* Wh1 = (const float*)d_in[5];
  const float* b1  = (const float*)d_in[6];

  char* ws = (char*)d_ws;
  size_t off = 0;
  unsigned* cnt  = (unsigned*)(ws + off); off += 256;
  __bf16* wxt0   = (__bf16*)(ws + off);   off += (size_t)G_ * K_ * 2;
  __bf16* wht0   = (__bf16*)(ws + off);   off += (size_t)G_ * K_ * 2;
  __bf16* wxt1   = (__bf16*)(ws + off);   off += (size_t)G_ * K_ * 2;
  __bf16* wht1   = (__bf16*)(ws + off);   off += (size_t)G_ * K_ * 2;
  __bf16* hbuf   = (__bf16*)(ws + off);   off += (size_t)2 * B_ * H_ * 2;
  float*  cst    = (float*)(ws + off);    off += (size_t)B_ * H_ * 4;
  __bf16* y0     = (__bf16*)(ws + off);   off += (size_t)M_ * H_ * 2;
  float*  xz     = (float*)(ws + off);    off += (size_t)M_ * G_ * 4;

  float* yout = (float*)d_out;
  float* hfin = yout + (size_t)M_ * H_;
  float* cfin = hfin + (size_t)2 * B_ * H_;

  // Weight conversion / transpose to bf16 [G_, K_]
  dim3 cvtg((K_ * G_) / 256), cvtb(256);
  k_wt_bf16<<<cvtg, cvtb, 0, stream>>>(Wx0, wxt0);
  k_wt_bf16<<<cvtg, cvtb, 0, stream>>>(Wh0, wht0);
  k_wt_bf16<<<cvtg, cvtb, 0, stream>>>(Wx1, wxt1);
  k_wt_bf16<<<cvtg, cvtb, 0, stream>>>(Wh1, wht1);

  dim3 gg(M_ / 64, G_ / 64), gb(128);
  dim3 rg(H_ / 16), rb(128);
  const size_t recur_lds =
      (size_t)64 * WSTRIDE * 2 + (size_t)4 * 64 * 16 * 4;  // ~81 KB

  // ---- Layer 0 ----
  hipMemsetAsync(hbuf, 0, (size_t)2 * B_ * H_ * 2, stream);
  hipMemsetAsync(cst,  0, (size_t)B_ * H_ * 4, stream);
  hipMemsetAsync(cnt,  0, 256, stream);
  k_gemm_xz<true><<<gg, gb, 0, stream>>>(x, wxt0, b0, xz);
  k_lstm_recur<false><<<rg, rb, recur_lds, stream>>>(xz, wht0, hbuf, cst, y0,
                                                     nullptr, hfin, cfin, cnt);

  // ---- Layer 1 ----
  hipMemsetAsync(hbuf, 0, (size_t)2 * B_ * H_ * 2, stream);
  hipMemsetAsync(cst,  0, (size_t)B_ * H_ * 4, stream);
  hipMemsetAsync(cnt,  0, 256, stream);
  k_gemm_xz<false><<<gg, gb, 0, stream>>>(y0, wxt1, b1, xz);
  k_lstm_recur<true><<<rg, rb, recur_lds, stream>>>(xz, wht1, hbuf, cst, nullptr,
                                                    yout,
                                                    hfin + (size_t)B_ * H_,
                                                    cfin + (size_t)B_ * H_, cnt);
}
